// AscendQwen3Attention_77455440216773
// MI455X (gfx1250) — matrix operand order repaired and checked by reading, not compiled, not run on hardware
//
#include <hip/hip_runtime.h>
#include <hip/hip_bf16.h>

typedef __attribute__((ext_vector_type(16))) _Float16 v16h;
typedef __attribute__((ext_vector_type(8)))  _Float16 v8h;
typedef __attribute__((ext_vector_type(4)))  _Float16 v4h;
typedef __attribute__((ext_vector_type(8)))  float    v8f;

constexpr int T_SEQ  = 2048;
constexpr int HIDDEN = 2048;
constexpr int HQ = 32, HK = 8, DH = 128;
constexpr int QKV_N = (HQ + 2 * HK) * DH;   // 6144
constexpr int ATTN_N = HQ * DH;             // 4096

// Padded LDS strides (halves). Odd multiple of 8 halves (16B): keeps v8h
// alignment for ds_load_b128 while decorrelating row-start banks.
constexpr int LDG = 40;    // GEMM tiles: 32 + 8
constexpr int LDK = 136;   // attention K tile: 128 + 8
constexpr int LDV = 72;    // attention V^T / P tiles: 64 + 8

union AFrag { v16h v; v8h p[2]; _Float16 h[16]; };

// A-matrix 16x32 f16 fragment from row-major [M][K] tile (ld = row stride).
// ISA 7.12.2: lanes0-15 M=lane, VGPR0-3=K0..7, VGPR4-7=K16..23;
//             lanes16-31 M=lane-16, K8..15 / K24..31.
__device__ inline v16h ldsAfrag(const _Float16* base, int ld) {
  int lane = threadIdx.x & 31;
  int m = lane & 15, hf = lane >> 4;
  const _Float16* row = base + m * ld;
  AFrag f;
  f.p[0] = *(const v8h*)(row + 8 * hf);
  f.p[1] = *(const v8h*)(row + 16 + 8 * hf);
  return f.v;
}

// B-matrix 32x16 f16 fragment from a TRANSPOSED [N][K] tile (ld = row stride).
// lanes0-15: N=lane, K0..7 / K16..23; lanes16-31: N=lane-16, K8..15 / K24..31.
__device__ inline v16h ldsBfragT(const _Float16* baseT, int ld) {
  int lane = threadIdx.x & 31;
  int n = lane & 15, hf = lane >> 4;
  const _Float16* row = baseT + n * ld;
  AFrag f;
  f.p[0] = *(const v8h*)(row + 8 * hf);
  f.p[1] = *(const v8h*)(row + 16 + 8 * hf);
  return f.v;
}

__device__ inline v8f wmma16(v16h a, v16h b, v8f c) {
  return __builtin_amdgcn_wmma_f32_16x16x32_f16(false, a, false, b,
                                                (short)0, c, false, false);
}

// ---------------------------------------------------------------------------
// Generic C = A(f32 or f16) * B(f32) GEMM, f16 WMMA, fp32 accumulate.
// Block tile 128x128, K step 32, 8 waves (256 thr), wave tile 32x64,
// double-buffered LDS (one barrier per K step, loads overlap WMMA).
// ---------------------------------------------------------------------------
template <typename TA>
__device__ inline void stage_tiles(const TA* __restrict__ A,
                                   const float* __restrict__ B,
                                   _Float16* __restrict__ sA,
                                   _Float16* __restrict__ sBt,
                                   int m0, int n0, int k0, int K, int N,
                                   int tid) {
  // A tile 128x32 (row-major, padded), 4 elems per chunk.
  for (int i = tid * 4; i < 128 * 32; i += 256 * 4) {
    int r = i >> 5, c = i & 31;
    const TA* src = A + (size_t)(m0 + r) * K + k0 + c;
    _Float16* dst = sA + r * LDG + c;
    if constexpr (sizeof(TA) == 4) {
      float4 f = *(const float4*)src;
      dst[0] = (_Float16)f.x; dst[1] = (_Float16)f.y;
      dst[2] = (_Float16)f.z; dst[3] = (_Float16)f.w;
    } else {
      *(v4h*)dst = *(const v4h*)src;
    }
  }
  // B tile 32x128 -> transposed [n][k] (coalesced global reads over n).
  for (int i = tid; i < 32 * 128; i += 256) {
    int k = i >> 7, n = i & 127;
    sBt[n * LDG + k] = (_Float16)B[(size_t)(k0 + k) * N + n0 + n];
  }
}

template <typename TA>
__global__ __launch_bounds__(256) void gemm_wmma(const TA* __restrict__ A,
                                                 const float* __restrict__ B,
                                                 float* __restrict__ C,
                                                 int M, int N, int K) {
  __shared__ _Float16 sA[2][128 * LDG];
  __shared__ _Float16 sBt[2][128 * LDG];
  const int m0 = blockIdx.y * 128, n0 = blockIdx.x * 128;
  const int tid = threadIdx.x;
  const int w = tid >> 5;
  const int wm = (w & 3) * 32;    // 4 wave rows  x 32
  const int wn = (w >> 2) * 64;   // 2 wave cols  x 64

  v8f acc[2][4];
  for (int i = 0; i < 2; i++)
    for (int j = 0; j < 4; j++)
      for (int r = 0; r < 8; r++) acc[i][j][r] = 0.f;

  stage_tiles(A, B, sA[0], sBt[0], m0, n0, 0, K, N, tid);
  __syncthreads();

  const int nk = K / 32;
  for (int kt = 0; kt < nk; kt++) {
    const int buf = kt & 1;
    if (kt + 1 < nk)   // prefetch next tile into the other buffer
      stage_tiles(A, B, sA[buf ^ 1], sBt[buf ^ 1], m0, n0, (kt + 1) * 32, K, N,
                  tid);
    v16h a0 = ldsAfrag(sA[buf] + (wm + 0) * LDG, LDG);
    v16h a1 = ldsAfrag(sA[buf] + (wm + 16) * LDG, LDG);
    for (int j = 0; j < 4; j++) {
      v16h bj = ldsBfragT(sBt[buf] + (wn + j * 16) * LDG, LDG);
      acc[0][j] = wmma16(a0, bj, acc[0][j]);
      acc[1][j] = wmma16(a1, bj, acc[1][j]);
    }
    __syncthreads();  // staging of buf^1 visible; buf free for next stage
  }

  const int lane = tid & 31, n = lane & 15, hf = lane >> 4;
  for (int i = 0; i < 2; i++)
    for (int j = 0; j < 4; j++) {
      int mb = m0 + wm + i * 16 + 8 * hf;   // VGPR r -> row mb + r
      int nb = n0 + wn + j * 16 + n;
      for (int r = 0; r < 8; r++)
        C[(size_t)(mb + r) * N + nb] = acc[i][j][r];
    }
}

// ---------------------------------------------------------------------------
// Per-head RMSNorm + 3-section MRoPE + f16 repack. One wave per (token, slot).
// slot 0..31 = Q heads, 32..39 = K heads, 40..47 = V heads (copy only).
// Q is pre-scaled by D^-1/2 so attention skips the softmax scale.
// ---------------------------------------------------------------------------
__global__ __launch_bounds__(256) void fuse_norm_rope(
    const float* __restrict__ qkv, const int* __restrict__ pos,
    const float* __restrict__ qw, const float* __restrict__ kw,
    _Float16* __restrict__ qb, _Float16* __restrict__ kb,
    _Float16* __restrict__ vb) {
  int wave = blockIdx.x * 8 + (threadIdx.x >> 5);
  int lane = threadIdx.x & 31;
  int t = wave / 48, slot = wave - t * 48;

  if (slot >= 40) {  // V: plain fp32 -> f16 copy
    int hv = slot - 40;
    const float* src = qkv + (size_t)t * QKV_N + (HQ + HK) * DH + hv * DH;
    _Float16* dst = vb + ((size_t)t * HK + hv) * DH;
    for (int i = lane; i < DH; i += 32) dst[i] = (_Float16)src[i];
    return;
  }
  const bool isq = slot < HQ;
  const int h = isq ? slot : slot - HQ;
  const float* src = qkv + (size_t)t * QKV_N + (isq ? h * DH : HQ * DH + h * DH);
  const float* wgt = isq ? qw : kw;
  _Float16* dst = isq ? (qb + ((size_t)t * HQ + h) * DH)
                      : (kb + ((size_t)t * HK + h) * DH);

  float a0 = src[lane], a1 = src[lane + 32];
  float b0 = src[lane + 64], b1 = src[lane + 96];
  float ss = a0 * a0 + a1 * a1 + b0 * b0 + b1 * b1;
  for (int off = 16; off; off >>= 1) ss += __shfl_xor(ss, off, 32);
  float rstd = rsqrtf(ss * (1.0f / 128.0f) + 1e-6f);
  float x0 = a0 * rstd * wgt[lane],      x1 = a1 * rstd * wgt[lane + 32];
  float y0 = b0 * rstd * wgt[lane + 64], y1 = b1 * rstd * wgt[lane + 96];

  // MRoPE: half-index j; sections (16,24,24) -> boundaries 16, 40.
  float c0, s0, c1, s1;
  {
    int j = lane;                               // 0..31
    int sect = (j < 16) ? 0 : 1;
    float p = (float)pos[sect * T_SEQ + t];
    float f = p * __expf(-(float)j * (13.815510557964274f / 64.0f));
    __sincosf(f, &s0, &c0);
  }
  {
    int j = lane + 32;                          // 32..63
    int sect = (j < 40) ? 1 : 2;
    float p = (float)pos[sect * T_SEQ + t];
    float f = p * __expf(-(float)j * (13.815510557964274f / 64.0f));
    __sincosf(f, &s1, &c1);
  }
  float scale = isq ? 0.08838834764831845f : 1.0f;  // D^-0.5 folded into Q
  dst[lane]      = (_Float16)((x0 * c0 - y0 * s0) * scale);
  dst[lane + 32] = (_Float16)((x1 * c1 - y1 * s1) * scale);
  dst[lane + 64] = (_Float16)((y0 * c0 + x0 * s0) * scale);
  dst[lane + 96] = (_Float16)((y1 * c1 + x1 * s1) * scale);
}

// ---------------------------------------------------------------------------
// Causal flash attention. Grid (T/128, HQ), 256 threads (8 waves).
// Wave w owns 16 Q rows; K/V tiles (64 kv rows) shared through LDS.
// S = Q*K^T via WMMA, online softmax in C-layout, P restaged through
// per-wave LDS into A-fragment layout, O += P*V via WMMA.
// ---------------------------------------------------------------------------
__global__ __launch_bounds__(256) void flash_attn(
    const _Float16* __restrict__ qb, const _Float16* __restrict__ kb,
    const _Float16* __restrict__ vb, _Float16* __restrict__ attn) {
  __shared__ _Float16 sK[64 * LDK];       // [kv][d]  (B-frags for S)
  __shared__ _Float16 sVt[128 * LDV];     // [d][kv]  (B-frags for PV)
  __shared__ _Float16 sP[8][16 * LDV];    // per-wave P tile [16][64]
  const int h = blockIdx.y;
  const int q0 = blockIdx.x * 128;
  const int hk = h >> 2;                  // GQA: 4 Q heads per KV head
  const int tid = threadIdx.x, w = tid >> 5, lane = tid & 31;
  const int m = lane & 15, hf = lane >> 4;
  const int qrb = q0 + w * 16;

  // Q A-fragments straight from global (f16, pre-scaled).
  v16h qf[4];
  for (int c = 0; c < 4; c++) {
    const _Float16* row = qb + ((size_t)(qrb + m) * HQ + h) * DH + c * 32;
    AFrag f;
    f.p[0] = *(const v8h*)(row + 8 * hf);
    f.p[1] = *(const v8h*)(row + 16 + 8 * hf);
    qf[c] = f.v;
  }

  v8f acc[8];
  for (int j = 0; j < 8; j++)
    for (int r = 0; r < 8; r++) acc[j][r] = 0.f;
  float mrow[8], lrow[8];
  for (int r = 0; r < 8; r++) { mrow[r] = -1e30f; lrow[r] = 0.f; }

  const int nblk = (q0 + 128) / 64;
  for (int kbi = 0; kbi < nblk; kbi++) {
    const int kv0 = kbi * 64;
    __syncthreads();  // previous iteration done reading K/V tiles
    for (int i = tid * 4; i < 64 * 128; i += 256 * 4) {
      int r = i >> 7, c = i & 127;
      *(v4h*)&sK[r * LDK + c] =
          *(const v4h*)&kb[((size_t)(kv0 + r) * HK + hk) * DH + c];
    }
    for (int i = tid; i < 64 * 128; i += 256) {
      int r = i >> 7, c = i & 127;
      sVt[c * LDV + r] = vb[((size_t)(kv0 + r) * HK + hk) * DH + c];
    }
    __syncthreads();

    // S tile 16x64 = 4 n-tiles, each accumulated over 4 d-chunks.
    v8f s[4];
    for (int j = 0; j < 4; j++) {
      v8f ca;
      for (int r = 0; r < 8; r++) ca[r] = 0.f;
      for (int c = 0; c < 4; c++)
        ca = wmma16(qf[c], ldsBfragT(sK + (j * 16) * LDK + c * 32, LDK), ca);
      s[j] = ca;
    }

    // Causal mask + row max (rows split across lane halves per C layout).
    float mnew[8];
    for (int r = 0; r < 8; r++) mnew[r] = mrow[r];
    for (int j = 0; j < 4; j++)
      for (int r = 0; r < 8; r++) {
        int qr = qrb + r + 8 * hf;
        int kc = kv0 + j * 16 + m;
        float v = (kc <= qr) ? s[j][r] : -1e30f;
        s[j][r] = v;
        mnew[r] = fmaxf(mnew[r], v);
      }
    for (int off = 1; off < 16; off <<= 1)
      for (int r = 0; r < 8; r++)
        mnew[r] = fmaxf(mnew[r], __shfl_xor(mnew[r], off, 32));

    float alpha[8], ladd[8];
    for (int r = 0; r < 8; r++) {
      alpha[r] = __expf(mrow[r] - mnew[r]);
      mrow[r] = mnew[r];
      ladd[r] = 0.f;
    }
    for (int j = 0; j < 4; j++)
      for (int r = 0; r < 8; r++) {
        float p = __expf(s[j][r] - mnew[r]);
        s[j][r] = p;
        ladd[r] += p;
      }
    for (int off = 1; off < 16; off <<= 1)
      for (int r = 0; r < 8; r++) ladd[r] += __shfl_xor(ladd[r], off, 32);
    for (int r = 0; r < 8; r++) lrow[r] = lrow[r] * alpha[r] + ladd[r];
    for (int j = 0; j < 8; j++)
      for (int r = 0; r < 8; r++) acc[j][r] *= alpha[r];

    // Restage P (C layout -> row-major LDS -> A fragments).
    _Float16* pw = &sP[w][0];
    for (int j = 0; j < 4; j++)
      for (int r = 0; r < 8; r++)
        pw[(r + 8 * hf) * LDV + j * 16 + m] = (_Float16)s[j][r];
    __syncthreads();

    // O(16x128) += P(16x64) * V(64x128): 8 d-tiles x 2 kv-chunks.
    for (int jd = 0; jd < 8; jd++)
      for (int kc2 = 0; kc2 < 2; kc2++)
        acc[jd] = wmma16(ldsAfrag(pw + kc2 * 32, LDV),
                         ldsBfragT(sVt + (jd * 16) * LDV + kc2 * 32, LDV),
                         acc[jd]);
  }

  for (int jd = 0; jd < 8; jd++)
    for (int r = 0; r < 8; r++) {
      int t = qrb + r + 8 * hf;
      float o = acc[jd][r] / lrow[r];
      attn[(size_t)t * ATTN_N + h * DH + jd * 16 + m] = (_Float16)o;
    }
}

// ---------------------------------------------------------------------------
extern "C" void kernel_launch(void* const* d_in, const int* in_sizes, int n_in,
                              void* d_out, int out_size, void* d_ws,
                              size_t ws_size, hipStream_t stream) {
  const float* hidden = (const float*)d_in[0];
  const int* positions = (const int*)d_in[1];
  const float* w_qkv = (const float*)d_in[2];
  const float* w_o = (const float*)d_in[3];
  const float* q_norm_w = (const float*)d_in[4];
  const float* k_norm_w = (const float*)d_in[5];
  float* out = (float*)d_out;

  char* ws = (char*)d_ws;
  size_t off = 0;
  float* qkv = (float*)(ws + off);        off += (size_t)T_SEQ * QKV_N * 4;   // 48 MB
  _Float16* qbuf = (_Float16*)(ws + off); off += (size_t)T_SEQ * HQ * DH * 2; // 16 MB
  _Float16* kbuf = (_Float16*)(ws + off); off += (size_t)T_SEQ * HK * DH * 2; //  4 MB
  _Float16* vbuf = (_Float16*)(ws + off); off += (size_t)T_SEQ * HK * DH * 2; //  4 MB
  _Float16* attn = (_Float16*)(ws + off);                                     // 16 MB

  // 1) QKV projection: [2048,2048] x [2048,6144]
  gemm_wmma<float><<<dim3(QKV_N / 128, T_SEQ / 128), 256, 0, stream>>>(
      hidden, w_qkv, qkv, T_SEQ, QKV_N, HIDDEN);

  // 2) RMSNorm + MRoPE + f16 repack (48 waves per token)
  fuse_norm_rope<<<T_SEQ * 48 / 8, 256, 0, stream>>>(
      qkv, positions, q_norm_w, k_norm_w, qbuf, kbuf, vbuf);

  // 3) Causal flash attention
  flash_attn<<<dim3(T_SEQ / 128, HQ), 256, 0, stream>>>(qbuf, kbuf, vbuf, attn);

  // 4) Output projection: [2048,4096] x [4096,2048]
  gemm_wmma<_Float16><<<dim3(HIDDEN / 128, T_SEQ / 128), 256, 0, stream>>>(
      attn, w_o, out, T_SEQ, HIDDEN, ATTN_N);
}
